// MultiHeadCausalAttention_28827820490878
// MI455X (gfx1250) — matrix operand order repaired and checked by reading, not compiled, not run on hardware
//
#include <hip/hip_runtime.h>

typedef __attribute__((ext_vector_type(16))) __bf16 v16bf;
typedef __attribute__((ext_vector_type(8)))  __bf16 v8bf;
typedef __attribute__((ext_vector_type(8)))  float  v8f;
typedef __attribute__((ext_vector_type(4)))  float  fl4;

#define DM     2048
#define TSEQ   2048
#define BATCH  4
#define NHEAD  16
#define HDIM   128
#define NTOK   (BATCH*TSEQ)
#define SCALE  45.254833995939045f   // sqrt(2048): faithful to the reference's buggy scaling

static __device__ __forceinline__ __bf16 f2bf(float f) { return (__bf16)f; }

// A-fragment (16x32 bf16, MxK). ISA 7.12.2: lane<16 -> K {0..7,16..23},
// lane>=16 -> K {8..15,24..31}; elems 0-7 = first chunk, 8-15 = second chunk.
static __device__ __forceinline__ v16bf ldAfrag(const __bf16* base, int pitch,
                                                int row, int kbase, int khalf) {
    const __bf16* p = base + row*pitch + kbase + khalf*8;
    v16bf f;
    ((v8bf*)&f)[0] = *(const v8bf*)(p);
    ((v8bf*)&f)[1] = *(const v8bf*)(p + 16);
    return f;
}

// B-fragment (32x16 bf16, KxN). Lane&15 = N column; K as A: lane<16 -> {0..7,16..23}, lane>=16 -> {8..15,24..31}.
static __device__ __forceinline__ v16bf ldBfrag(const __bf16* base, int pitch,
                                                int krow, int nbase) {
    const int l = krow & 31;
    const __bf16* p = base + ((krow & ~31) + (l >> 4)*8)*pitch + nbase + (l & 15);
    v16bf f;
    #pragma unroll
    for (int s = 0; s < 8; s++) { f[s] = p[s*pitch]; f[8+s] = p[(16+s)*pitch]; }
    return f;
}

// C = A @ Bw. A is MxK (fp32 or bf16), Bw is KxN fp32 (converted to bf16 in LDS),
// C is MxN (fp32 or bf16). f32 accumulation via v_wmma_f32_16x16x32_bf16.
// BM=128, BN=128, BK=32; 8 waves, each owns 32x64 (8 accumulators, 8 WMMAs/step).
// Double-buffered LDS + register prefetch: one barrier per K-step.
template<typename AT, typename CT>
__global__ __launch_bounds__(256)
void gemm_kernel(const AT* __restrict__ A, const float* __restrict__ Bw,
                 CT* __restrict__ C, int M, int N, int K) {
    constexpr int BM = 128, BN = 128, BK = 32;
    constexpr int AP = 40, BP = 136;           // padded pitches (16B-aligned rows)
    constexpr int ASZ = BM*AP, BSZ = BK*BP;
    __shared__ __attribute__((aligned(16))) __bf16 As[2*ASZ];
    __shared__ __attribute__((aligned(16))) __bf16 Bs[2*BSZ];

    const int tid  = threadIdx.x;
    const int lane = tid & 31;
    const int wid  = tid >> 5;
    const int wm   = wid & 3;                  // 4 wave-rows of 32
    const int wn   = wid >> 2;                 // 2 wave-cols of 64
    const int m0   = blockIdx.y * BM;
    const int n0   = blockIdx.x * BN;

    v8f acc[2][4] = {};

    fl4  apre_f[4];                            // fp32 A prefetch (16 floats/thread)
    v8bf apre_b[2];                            // bf16 A prefetch (16 bf16/thread)
    fl4  bpre[4];                              // B prefetch (16 floats/thread)

    // ---- prefetch helpers (lambdas keep index math in one place) ----
    auto loadA = [&](int k0) {
        if constexpr (sizeof(AT) == 4) {
            #pragma unroll
            for (int i = 0; i < 4; i++) {
                int lin = tid + i*256, row = lin >> 3, cg = lin & 7;   // 1024 fl4 chunks
                apre_f[i] = *(const fl4*)((const float*)A + (size_t)(m0+row)*K + k0 + cg*4);
            }
        } else {
            #pragma unroll
            for (int i = 0; i < 2; i++) {
                int lin = tid + i*256, row = lin >> 2, cg = lin & 3;   // 512 v8bf chunks
                apre_b[i] = *(const v8bf*)((const __bf16*)A + (size_t)(m0+row)*K + k0 + cg*8);
            }
        }
    };
    auto loadB = [&](int k0) {
        #pragma unroll
        for (int i = 0; i < 4; i++) {
            int lin = tid + i*256, row = lin >> 5, cg = lin & 31;      // 1024 fl4 chunks
            bpre[i] = *(const fl4*)(Bw + (size_t)(k0+row)*N + n0 + cg*4);
        }
    };
    auto storeTiles = [&](int buf) {
        __bf16* as = &As[buf*ASZ];
        __bf16* bs = &Bs[buf*BSZ];
        if constexpr (sizeof(AT) == 4) {
            #pragma unroll
            for (int i = 0; i < 4; i++) {
                int lin = tid + i*256, row = lin >> 3, cg = lin & 7;
                __bf16* dst = &as[row*AP + cg*4];
                dst[0]=f2bf(apre_f[i].x); dst[1]=f2bf(apre_f[i].y);
                dst[2]=f2bf(apre_f[i].z); dst[3]=f2bf(apre_f[i].w);
            }
        } else {
            #pragma unroll
            for (int i = 0; i < 2; i++) {
                int lin = tid + i*256, row = lin >> 2, cg = lin & 3;
                *(v8bf*)&as[row*AP + cg*8] = apre_b[i];
            }
        }
        #pragma unroll
        for (int i = 0; i < 4; i++) {
            int lin = tid + i*256, row = lin >> 5, cg = lin & 31;
            __bf16* dst = &bs[row*BP + cg*4];
            dst[0]=f2bf(bpre[i].x); dst[1]=f2bf(bpre[i].y);
            dst[2]=f2bf(bpre[i].z); dst[3]=f2bf(bpre[i].w);
        }
    };

    // ---- prologue: fill buffer 0 ----
    loadA(0); loadB(0);
    storeTiles(0);
    __syncthreads();

    int cur = 0;
    for (int k0 = BK; k0 <= K; k0 += BK) {
        const bool has_next = (k0 < K);
        if (has_next) { loadA(k0); loadB(k0); }  // global -> regs, overlaps compute

        const __bf16* as = &As[cur*ASZ];
        const __bf16* bs = &Bs[cur*BSZ];
        v16bf af[2], bfr[4];
        #pragma unroll
        for (int i = 0; i < 2; i++)
            af[i] = ldAfrag(as, AP, wm*32 + i*16 + (lane & 15), 0, lane >> 4);
        #pragma unroll
        for (int j = 0; j < 4; j++)
            bfr[j] = ldBfrag(bs, BP, lane, wn*64 + j*16);
        #pragma unroll
        for (int i = 0; i < 2; i++)
            #pragma unroll
            for (int j = 0; j < 4; j++)
                acc[i][j] = __builtin_amdgcn_wmma_f32_16x16x32_bf16(
                    false, af[i], false, bfr[j], (short)0, acc[i][j], false, false);

        if (has_next) {
            storeTiles(cur ^ 1);               // regs -> other buffer
            __syncthreads();                   // single barrier per K-step
            cur ^= 1;
        }
    }

    // ---- epilogue: C VGPR r -> M = r + (lane>>4)*8, N = lane&15 ----
    const int rbase = m0 + wm*32, cbase = n0 + wn*64;
    #pragma unroll
    for (int i = 0; i < 2; i++)
        #pragma unroll
        for (int j = 0; j < 4; j++)
            #pragma unroll
            for (int r = 0; r < 8; r++) {
                int row = rbase + i*16 + r + ((lane >> 4) << 3);
                int col = cbase + j*16 + (lane & 15);
                float v = acc[i][j][r];
                if constexpr (sizeof(CT) == 4) C[(size_t)row*N + col] = v;
                else                           C[(size_t)row*N + col] = f2bf(v);
            }
}

// Flash-style causal attention over 64-query blocks; S and P@V via WMMA bf16.
__global__ __launch_bounds__(256)
void attn_kernel(const __bf16* __restrict__ Q, const __bf16* __restrict__ Kg,
                 const __bf16* __restrict__ Vg, __bf16* __restrict__ O) {
    constexpr int BR = 64, BC = 64;
    constexpr int QP = 136, KP = 72, VP = 136, SP = 65, PP = 72;
    __shared__ __attribute__((aligned(16))) __bf16 Qs[BR*QP];
    __shared__ __attribute__((aligned(16))) __bf16 Kt[HDIM*KP];   // transposed: [d][s]
    __shared__ __attribute__((aligned(16))) __bf16 Vs[BC*VP];
    __shared__ __attribute__((aligned(16))) float  Sb[BR*SP];
    __shared__ __attribute__((aligned(16))) __bf16 Pb[BR*PP];
    __shared__ __attribute__((aligned(16))) float  Of[BR*HDIM];
    __shared__ float mrow[BR], lrow[BR], arow[BR];

    const int tid  = threadIdx.x;
    const int lane = tid & 31;
    const int wid  = tid >> 5;
    const int wm   = wid & 3;    // 4 wave-rows of 16 queries
    const int wn   = wid >> 2;   // 2 wave-cols
    const int q0   = blockIdx.x * BR;
    const int h    = blockIdx.y;
    const int b    = blockIdx.z;
    const size_t base = (size_t)b*TSEQ*DM + (size_t)h*HDIM;

    // ---- load Q tile, init O/m/l ----
    #pragma unroll
    for (int i = 0; i < 4; i++) {
        int lin = tid + i*256;               // 1024 v8bf chunks
        int row = lin >> 4, cg = lin & 15;
        *(v8bf*)&Qs[row*QP + cg*8] = *(const v8bf*)(Q + base + (size_t)(q0+row)*DM + cg*8);
    }
    #pragma unroll
    for (int i = 0; i < 32; i++) Of[tid*32 + i] = 0.f;
    if (tid < BR) { mrow[tid] = -3.0e38f; lrow[tid] = 0.f; }

    const int nkb = q0/BC + 1;               // causal: only key blocks <= query block
    for (int kb = 0; kb < nkb; kb++) {
        const int k0 = kb*BC;
        __syncthreads();
        // ---- stage K (transposed) and V tiles ----
        #pragma unroll
        for (int i = 0; i < 4; i++) {
            int lin = tid + i*256;
            int row = lin >> 4, cg = lin & 15;
            v8bf kv = *(const v8bf*)(Kg + base + (size_t)(k0+row)*DM + cg*8);
            #pragma unroll
            for (int e = 0; e < 8; e++) Kt[(cg*8+e)*KP + row] = kv[e];
            *(v8bf*)&Vs[row*VP + cg*8] = *(const v8bf*)(Vg + base + (size_t)(k0+row)*DM + cg*8);
        }
        __syncthreads();
        // ---- S = Q @ K^T (scaled) ----
        v8f sacc[2] = {};
        #pragma unroll
        for (int kc = 0; kc < 4; kc++) {     // head_dim 128 = 4 x K32
            v16bf a = ldAfrag(Qs, QP, wm*16 + (lane & 15), kc*32, lane >> 4);
            #pragma unroll
            for (int j = 0; j < 2; j++) {
                v16bf bb = ldBfrag(Kt, KP, kc*32 + lane, wn*32 + j*16);
                sacc[j] = __builtin_amdgcn_wmma_f32_16x16x32_bf16(
                    false, a, false, bb, (short)0, sacc[j], false, false);
            }
        }
        #pragma unroll
        for (int j = 0; j < 2; j++)
            #pragma unroll
            for (int r = 0; r < 8; r++)
                Sb[(wm*16 + r + ((lane >> 4) << 3))*SP + wn*32 + j*16 + (lane & 15)]
                    = sacc[j][r] * SCALE;
        __syncthreads();
        // ---- online softmax per row (64 worker threads) ----
        if (tid < BR) {
            int row = tid, qi = q0 + row;
            int climit = qi - k0 + 1; if (climit > BC) climit = BC;
            float mo = mrow[row], mx = mo;
            for (int c = 0; c < climit; c++) mx = fmaxf(mx, Sb[row*SP + c]);
            float alpha = __expf(mo - mx), sum = 0.f;
            for (int c = 0; c < BC; c++) {
                float p = (c < climit) ? __expf(Sb[row*SP + c] - mx) : 0.f;
                sum += p;
                Pb[row*PP + c] = f2bf(p);
            }
            mrow[row] = mx; lrow[row] = lrow[row]*alpha + sum; arow[row] = alpha;
        }
        __syncthreads();
        // ---- rescale running O by alpha ----
        {
            int row = tid >> 2, c0 = (tid & 3)*32;
            float al = arow[row];
            #pragma unroll
            for (int i = 0; i < 32; i++) Of[row*HDIM + c0 + i] *= al;
        }
        __syncthreads();
        // ---- O += P @ V ----
        v8f oacc[4] = {};
        #pragma unroll
        for (int kc = 0; kc < 2; kc++) {     // 64 keys = 2 x K32
            v16bf a = ldAfrag(Pb, PP, wm*16 + (lane & 15), kc*32, lane >> 4);
            #pragma unroll
            for (int j = 0; j < 4; j++) {
                v16bf bb = ldBfrag(Vs, VP, kc*32 + lane, wn*64 + j*16);
                oacc[j] = __builtin_amdgcn_wmma_f32_16x16x32_bf16(
                    false, a, false, bb, (short)0, oacc[j], false, false);
            }
        }
        #pragma unroll
        for (int j = 0; j < 4; j++)
            #pragma unroll
            for (int r = 0; r < 8; r++)
                Of[(wm*16 + r + ((lane >> 4) << 3))*HDIM + wn*64 + j*16 + (lane & 15)]
                    += oacc[j][r];
    }
    __syncthreads();
    // ---- normalize and write attn output (bf16) ----
    {
        int row = tid >> 2, c0 = (tid & 3)*32;
        float inv = 1.0f / lrow[row];
        #pragma unroll
        for (int i = 0; i < 32; i++)
            O[base + (size_t)(q0+row)*DM + c0 + i] = f2bf(Of[row*HDIM + c0 + i] * inv);
    }
}

extern "C" void kernel_launch(void* const* d_in, const int* in_sizes, int n_in,
                              void* d_out, int out_size, void* d_ws, size_t ws_size,
                              hipStream_t stream) {
    (void)in_sizes; (void)n_in; (void)out_size; (void)ws_size;
    const float* x  = (const float*)d_in[0];
    const float* Wq = (const float*)d_in[1];
    const float* Wk = (const float*)d_in[2];
    const float* Wv = (const float*)d_in[3];
    const float* Wo = (const float*)d_in[4];
    float* out = (float*)d_out;

    const size_t nQ = (size_t)NTOK * DM;          // 16.7M elems (bf16) each
    __bf16* qb = (__bf16*)d_ws;                   // 32 MiB
    __bf16* kb = qb + nQ;                         // 32 MiB
    __bf16* vb = kb + nQ;                         // 32 MiB
    __bf16* ab = vb + nQ;                         // 32 MiB  (128 MiB total)

    dim3 blk(256);
    dim3 gg(DM/128, NTOK/128);                    // (N/BN, M/BM)
    gemm_kernel<float,  __bf16><<<gg, blk, 0, stream>>>(x,  Wq, qb, NTOK, DM, DM);
    gemm_kernel<float,  __bf16><<<gg, blk, 0, stream>>>(x,  Wk, kb, NTOK, DM, DM);
    gemm_kernel<float,  __bf16><<<gg, blk, 0, stream>>>(x,  Wv, vb, NTOK, DM, DM);
    attn_kernel<<<dim3(TSEQ/64, NHEAD, BATCH), blk, 0, stream>>>(qb, kb, vb, ab);
    gemm_kernel<__bf16, float ><<<gg, blk, 0, stream>>>(ab, Wo, out, NTOK, DM, DM);
}
